// FSAS_78597901517111
// MI455X (gfx1250) — matrix operand order repaired and checked by reading, not compiled, not run on hardware
//
#include <hip/hip_runtime.h>
#include <hip/hip_bf16.h>

typedef __attribute__((ext_vector_type(16))) _Float16 v16h;
typedef __attribute__((ext_vector_type(8)))  _Float16 v8h;
typedef __attribute__((ext_vector_type(8)))  float    v8f;

union F16Frag { v16h v; v8h h[2]; };

// ---- problem constants ----
constexpr int M1   = 4096;   // patches (4 * 32 * 32)
constexpr int K1   = 1024;   // 8*8*16
constexpr int FEAT = 4096;
constexpr int NQKV = 12288;  // q|k|v fused
constexpr int K2   = 4096;   // 8*8*64
constexpr int N3   = 1024;   // 16*8*8

// ---- workspace layout (bytes) ----
constexpr size_t OFF_A1   = 0;
constexpr size_t SZ_A1    = (size_t)M1 * K1 * 2;
constexpr size_t OFF_B1   = OFF_A1 + SZ_A1;
constexpr size_t SZ_B1    = (size_t)K1 * NQKV * 2;
constexpr size_t OFF_C1   = OFF_B1 + SZ_B1;
constexpr size_t SZ_C1    = (size_t)M1 * NQKV * 2;
constexpr size_t OFF_BIA1 = OFF_C1 + SZ_C1;
constexpr size_t SZ_BIA1  = (size_t)NQKV * 4;
constexpr size_t OFF_A2   = OFF_BIA1 + SZ_BIA1;
constexpr size_t SZ_A2    = (size_t)M1 * K2 * 2;
constexpr size_t OFF_B3   = OFF_A2 + SZ_A2;
constexpr size_t SZ_B3    = (size_t)K2 * N3 * 2;
constexpr size_t OFF_Y    = OFF_B3 + SZ_B3;

// ---- CDNA5 async global->LDS path via inline asm (bypasses clang builtin
// whose signature differs between toolchains). Device-side gfx1250 only. ----
#if defined(__HIP_DEVICE_COMPILE__) && defined(__gfx1250__)
#define HAVE_ASYNC_LDS 1
#else
#define HAVE_ASYNC_LDS 0
#endif

__device__ __forceinline__ void async_copy_b128(const _Float16* g, _Float16* l) {
#if HAVE_ASYNC_LDS
    // VDST = LDS byte address (low 32 bits of flat shared pointer),
    // VADDR = 64-bit global address, no SADDR.
    const unsigned int       loff  = (unsigned int)(uintptr_t)l;
    const unsigned long long gaddr = (unsigned long long)(uintptr_t)g;
    asm volatile("global_load_async_to_lds_b128 %0, %1, off"
                 :: "v"(loff), "v"(gaddr)
                 : "memory");
#else
    *reinterpret_cast<v8h*>(l) = *reinterpret_cast<const v8h*>(g);
#endif
}

__device__ __forceinline__ void wait_async_lds() {
#if HAVE_ASYNC_LDS
    asm volatile("s_wait_asynccnt 0x0" ::: "memory");
#endif
}

// =====================================================================
// Pack kernels (f32 -> f16 with layout changes)
// =====================================================================
__global__ void pack_A1_kernel(const float* __restrict__ x, _Float16* __restrict__ A1) {
    int gid = blockIdx.x * 256 + threadIdx.x;          // 4096*1024 exact
    int m = gid >> 10, k = gid & 1023;
    int b = m >> 10, hh = (m >> 5) & 31, ww = m & 31;
    int p0 = k >> 7, p1 = (k >> 4) & 7, ci = k & 15;
    int h = hh * 8 + p0, w = ww * 8 + p1;
    A1[gid] = (_Float16)x[(((size_t)b * 256 + h) * 256 + w) * 16 + ci];
}

__global__ void pack_B1_kernel(const float* __restrict__ Wq, const float* __restrict__ Wk,
                               const float* __restrict__ Wv, _Float16* __restrict__ B1) {
    int n  = blockIdx.x * 256 + threadIdx.x;           // grid.x = 48
    int kk = blockIdx.y;                               // 0..1023
    int sel = n >> 12, nl = n & 4095;
    const float* Wsrc = (sel == 0) ? Wq : ((sel == 1) ? Wk : Wv);
    B1[(size_t)kk * NQKV + n] = (_Float16)Wsrc[(size_t)kk * FEAT + nl];
}

__global__ void pack_bias1_kernel(const float* __restrict__ bq, const float* __restrict__ bk,
                                  const float* __restrict__ bv, float* __restrict__ biasc) {
    int n = blockIdx.x * 256 + threadIdx.x;
    if (n >= NQKV) return;
    int sel = n >> 12, nl = n & 4095;
    biasc[n] = (sel == 0) ? bq[nl] : ((sel == 1) ? bk[nl] : bv[nl]);
}

__global__ void cvt_f2h_kernel(const float* __restrict__ src, _Float16* __restrict__ dst, int n) {
    int gid = blockIdx.x * 256 + threadIdx.x;
    if (gid < n) dst[gid] = (_Float16)src[gid];
}

// =====================================================================
// WMMA GEMM: C[MxN] = A[MxK] @ B[KxN] + bias[n]
// 256 threads = 8 waves as 2(M) x 4(N); block tile 128x128, BK=32.
// Each wave computes 64x32 = 4x2 v_wmma_f32_16x16x32_f16 per K-step.
// Double-buffered LDS; tiles staged with global_load_async_to_lds_b128.
// =====================================================================
template <bool OUT_HALF>
__global__ __launch_bounds__(256, 1)
void gemm_wmma_kernel(const _Float16* __restrict__ A, const _Float16* __restrict__ B,
                      const float* __restrict__ bias, void* __restrict__ Cout,
                      int M, int N, int K) {
    __shared__ __align__(16) _Float16 As[2][128][40];    // 128x32 + pad
    __shared__ __align__(16) _Float16 Bs[2][32][136];    // 32x128 + pad

    const int tid  = threadIdx.x;
    const int lane = tid & 31;
    const int wave = tid >> 5;
    const int wm = wave & 1;                  // 0..1  (64 rows each)
    const int wn = wave >> 1;                 // 0..3  (32 cols each)
    const int blockM = blockIdx.y * 128;
    const int blockN = blockIdx.x * 128;

    const v8f vzero = {0.f, 0.f, 0.f, 0.f, 0.f, 0.f, 0.f, 0.f};
    v8f acc[4][2];
#pragma unroll
    for (int i = 0; i < 4; i++) { acc[i][0] = vzero; acc[i][1] = vzero; }

    // tile-staging assignments (each thread moves 16+16 halves)
    const int arow = tid >> 1;                // 0..127
    const int aseg = (tid & 1) * 16;          // 0 | 16
    const int brow = tid >> 3;                // 0..31
    const int bseg = (tid & 7) * 16;          // 0..112

    // wave32 WMMA A-fragment addressing
    const int r    = lane & 15;
    const int koff = (lane >> 4) * 8;         // 0 | 8

    auto load_tile = [&](int k0, int buf) {
        const _Float16* ga = A + (size_t)(blockM + arow) * K + k0 + aseg;
        const _Float16* gb = B + (size_t)(k0 + brow) * N + blockN + bseg;
        _Float16* la = &As[buf][arow][aseg];
        _Float16* lb = &Bs[buf][brow][bseg];
        async_copy_b128(ga,     la);
        async_copy_b128(ga + 8, la + 8);
        async_copy_b128(gb,     lb);
        async_copy_b128(gb + 8, lb + 8);
    };

    const int nk = K >> 5;                    // K / 32 (K is 1024 or 4096)

    load_tile(0, 0);
    wait_async_lds();
    __syncthreads();

    for (int t = 0; t < nk; ++t) {
        const int cur = t & 1;
        if (t + 1 < nk) load_tile(t + 1, cur ^ 1);   // overlap with compute

        F16Frag afrag[4], bfrag[2];
#pragma unroll
        for (int i = 0; i < 4; i++) {
            // A 16x32: lanes 0-15 row r K{0..7,16..23}; lanes 16-31 row r K{8..15,24..31}
            const _Float16* p = &As[cur][wm * 64 + i * 16 + r][koff];
            afrag[i].h[0] = *reinterpret_cast<const v8h*>(p);
            afrag[i].h[1] = *reinterpret_cast<const v8h*>(p + 16);
        }
#pragma unroll
        for (int j = 0; j < 2; j++) {
            // B 32x16: lanes 0-15 col r K{0..7,16..23}; lanes 16-31 col r K{8..15,24..31}
#pragma unroll
            for (int s = 0; s < 16; s++)
                bfrag[j].v[s] = Bs[cur][koff + (s & 7) + (s >> 3) * 16][wn * 32 + j * 16 + r];
        }
#pragma unroll
        for (int i = 0; i < 4; i++)
#pragma unroll
            for (int j = 0; j < 2; j++)
                acc[i][j] = __builtin_amdgcn_wmma_f32_16x16x32_f16(
                    false, afrag[i].v, false, bfrag[j].v, (short)0, acc[i][j], false, false);

        wait_async_lds();     // all of this wave's async fills of buf cur^1 done
        __syncthreads();      // and all waves done reading buf cur
    }

    // epilogue: C layout — VGPR e: M=e (lanes 0-15) / M=e+8 (lanes 16-31), N=lane&15
    const int nbase = blockN + wn * 32 + (lane & 15);
    const int mtop  = blockM + wm * 64 + ((lane >> 4) << 3);
#pragma unroll
    for (int i = 0; i < 4; i++) {
#pragma unroll
        for (int j = 0; j < 2; j++) {
            const int n = nbase + j * 16;
            const float bvv = bias ? bias[n] : 0.0f;
#pragma unroll
            for (int e = 0; e < 8; e++) {
                const int m = mtop + i * 16 + e;
                const float val = acc[i][j][e] + bvv;
                if (OUT_HALF) ((_Float16*)Cout)[(size_t)m * N + n] = (_Float16)val;
                else          ((float*)Cout)[(size_t)m * N + n]    = val;
            }
        }
    }
}

// =====================================================================
// Stage 2: per-(patch m, head) 8x8 circular conv (== irfft2(rfft2(q)*rfft2(k))),
// LayerNorm over 16 channels (thread-local), multiply by v, emit im2col A2 (f16).
// C1 feature layout: n = sel*4096 + (c*64 + p0*8 + p1)*4 + head
// A2 layout: A2[m][ (p0*8+p1)*64 + head*16 + c ]
// =====================================================================
__global__ __launch_bounds__(64)
void freq_attn_ln_kernel(const _Float16* __restrict__ C1,
                         const float* __restrict__ ln_scale,
                         const float* __restrict__ ln_bias,
                         _Float16* __restrict__ A2) {
    __shared__ float qs[16][64];
    __shared__ float ks[16][64];

    const int t    = threadIdx.x;        // p0*8+p1
    const int gid  = blockIdx.x;         // m*4 + head
    const int head = gid & 3;
    const int m    = gid >> 2;
    const size_t base = (size_t)m * NQKV;

#pragma unroll
    for (int c = 0; c < 16; c++) {
        const int idx = ((c * 64 + t) << 2) + head;
        qs[c][t] = (float)C1[base + idx];
        ks[c][t] = (float)C1[base + 4096 + idx];
    }
    __syncthreads();

    const int mo = t >> 3, no = t & 7;
    float outc[16];
#pragma unroll
    for (int c = 0; c < 16; c++) outc[c] = 0.f;

    for (int i = 0; i < 8; i++) {
        const int mi = (mo - i) & 7;
#pragma unroll
        for (int j = 0; j < 8; j++) {
            const int nj   = (no - j) & 7;
            const int qidx = i * 8 + j;
            const int kidx = mi * 8 + nj;
#pragma unroll
            for (int c = 0; c < 16; c++)
                outc[c] = fmaf(qs[c][qidx], ks[c][kidx], outc[c]);
        }
    }

    float mu = 0.f;
#pragma unroll
    for (int c = 0; c < 16; c++) mu += outc[c];
    mu *= (1.0f / 16.0f);
    float var = 0.f;
#pragma unroll
    for (int c = 0; c < 16; c++) { const float d = outc[c] - mu; var += d * d; }
    var *= (1.0f / 16.0f);
    const float rs = rsqrtf(var + 1e-6f);

    _Float16* dst = A2 + (size_t)m * K2 + t * 64 + head * 16;
#pragma unroll
    for (int c = 0; c < 16; c++) {
        const float nrm = (outc[c] - mu) * rs * ln_scale[c] + ln_bias[c];
        const float vv  = (float)C1[base + 8192 + ((c * 64 + t) << 2) + head];
        dst[c] = (_Float16)(vv * nrm);
    }
}

// =====================================================================
// Scatter Y[m][n] (n = c*64 + p0*8 + p1) -> out NHWC [4,256,256,16]
// =====================================================================
__global__ void scatter_out_kernel(const float* __restrict__ Y, float* __restrict__ out) {
    int gid = blockIdx.x * 256 + threadIdx.x;
    int m = gid >> 10, n = gid & 1023;
    int b = m >> 10, hh = (m >> 5) & 31, ww = m & 31;
    int c = n >> 6, p0 = (n >> 3) & 7, p1 = n & 7;
    out[(((size_t)b * 256 + hh * 8 + p0) * 256 + (ww * 8 + p1)) * 16 + c] = Y[gid];
}

// =====================================================================
extern "C" void kernel_launch(void* const* d_in, const int* in_sizes, int n_in,
                              void* d_out, int out_size, void* d_ws, size_t ws_size,
                              hipStream_t stream) {
    const float* x    = (const float*)d_in[0];
    const float* Wq   = (const float*)d_in[1];
    const float* bq   = (const float*)d_in[2];
    const float* Wk   = (const float*)d_in[3];
    const float* bk   = (const float*)d_in[4];
    const float* Wv   = (const float*)d_in[5];
    const float* bv   = (const float*)d_in[6];
    const float* lns  = (const float*)d_in[7];
    const float* lnb  = (const float*)d_in[8];
    const float* Wout = (const float*)d_in[9];
    const float* bout = (const float*)d_in[10];
    float* out = (float*)d_out;

    char* ws = (char*)d_ws;
    _Float16* A1    = (_Float16*)(ws + OFF_A1);
    _Float16* B1    = (_Float16*)(ws + OFF_B1);
    _Float16* C1    = (_Float16*)(ws + OFF_C1);
    float*    bias1 = (float*)   (ws + OFF_BIA1);
    _Float16* A2    = (_Float16*)(ws + OFF_A2);
    _Float16* B3    = (_Float16*)(ws + OFF_B3);
    float*    Y     = (float*)   (ws + OFF_Y);

    pack_A1_kernel<<<dim3((M1 * K1) / 256), dim3(256), 0, stream>>>(x, A1);
    pack_B1_kernel<<<dim3(NQKV / 256, K1), dim3(256), 0, stream>>>(Wq, Wk, Wv, B1);
    pack_bias1_kernel<<<dim3((NQKV + 255) / 256), dim3(256), 0, stream>>>(bq, bk, bv, bias1);
    cvt_f2h_kernel<<<dim3((K2 * N3) / 256), dim3(256), 0, stream>>>(Wout, B3, K2 * N3);

    // GEMM 1: QKV = A1[4096x1024] @ B1[1024x12288] + bias1 (f16 out)
    gemm_wmma_kernel<true><<<dim3(NQKV / 128, M1 / 128), dim3(256), 0, stream>>>(
        A1, B1, bias1, (void*)C1, M1, NQKV, K1);

    // Stage 2: circular-conv "attention" + LN + v*, emits A2 (f16)
    freq_attn_ln_kernel<<<dim3(M1 * 4), dim3(64), 0, stream>>>(C1, lns, lnb, A2);

    // GEMM 3: Y = A2[4096x4096] @ B3[4096x1024] + bout (f32 out)
    gemm_wmma_kernel<false><<<dim3(N3 / 128, M1 / 128), dim3(256), 0, stream>>>(
        A2, B3, bout, (void*)Y, M1, N3, K2);

    scatter_out_kernel<<<dim3((M1 * N3) / 256), dim3(256), 0, stream>>>(Y, out);
}
